// AspectSent_43765716746611
// MI455X (gfx1250) — matrix operand order repaired and checked by reading, not compiled, not run on hardware
//
#include <hip/hip_runtime.h>
#include <stdint.h>

// ---------------- model constants ----------------
constexpr int Bc   = 128;
constexpr int Lc   = 128;
constexpr int Dc   = 300;
constexpr int MDc  = 50;
constexpr int GINc = 350;          // D + MD
constexpr int GINP = 352;          // padded to multiple of 32 for WMMA K
constexpr int HDc  = 256;          // per-direction hidden
constexpr int Hc   = 512;          // 2*HD
constexpr int Rc   = 128;
constexpr int Tc   = 4;
constexpr int G3   = 768;          // 3*HD
constexpr int BL   = Bc * Lc;      // 16384

// LDS B-tile row stride (ushorts): 32 data + 8 pad -> 80B rows (20 banks, odd,
// conflict-free for the 16-lane fragment reads, still 16B aligned)
constexpr int BT_RS = 40;

// ---------------- WMMA plumbing ----------------
typedef __attribute__((ext_vector_type(16))) __bf16 v16bf;
typedef __attribute__((ext_vector_type(8)))  float  v8f;
typedef __attribute__((ext_vector_type(4)))  unsigned int v4u;
typedef __attribute__((ext_vector_type(8)))  int  v8i_;
typedef __attribute__((ext_vector_type(4)))  int  v4i_;

union FragU { v16bf v; uint4 q[2]; };

#if defined(__has_builtin)
#if __has_builtin(__builtin_amdgcn_tensor_load_to_lds) && __has_builtin(__builtin_amdgcn_s_wait_tensorcnt)
#define HAVE_TDM 1
#endif
#endif
#ifndef HAVE_TDM
#define HAVE_TDM 0
#endif

__device__ __forceinline__ unsigned short f32_to_bf16(float f) {
    unsigned int u = __float_as_uint(f);
    unsigned int r = (u + 0x7FFFu + ((u >> 16) & 1u)) >> 16;   // RNE
    return (unsigned short)r;
}

// A fragment 16x32 bf16 from row-major [M][ld] matrix.
// lanes 0-15: rows m0..m0+15, K = k0+0..15
// lanes 16-31: same rows,      K = k0+16..31
__device__ __forceinline__ v16bf load_frag_a(const unsigned short* X, int ld,
                                             int m0, int k0, int lane) {
    int row = m0 + (lane & 15);
    int kb  = k0 + ((lane >> 4) << 4);
    const unsigned short* p = X + (size_t)row * ld + kb;
    FragU f;
    f.q[0] = *reinterpret_cast<const uint4*>(p);        // K kb..kb+7
    f.q[1] = *reinterpret_cast<const uint4*>(p + 8);    // K kb+8..kb+15
    return f.v;
}

// B fragment 32x16 bf16 from row-major [N][ld] weight (B(k,n) = W[n][k]).
// lanes 0-15: col n0+lane, K = k0+0..15 ; lanes 16-31: K = k0+16..31
__device__ __forceinline__ v16bf load_frag_b(const unsigned short* W, int ld,
                                             int n0, int k0, int lane) {
    int col = n0 + (lane & 15);
    int kb  = k0 + ((lane >> 4) << 4);
    const unsigned short* p = W + (size_t)col * ld + kb;
    FragU f;
    f.q[0] = *reinterpret_cast<const uint4*>(p);
    f.q[1] = *reinterpret_cast<const uint4*>(p + 8);
    return f.v;
}

// B fragment from an LDS-staged tile [128][BT_RS] (32 valid K per row).
__device__ __forceinline__ v16bf load_frag_b_lds(const unsigned short* T,
                                                 int n0rel, int lane) {
    int r  = n0rel + (lane & 15);
    int kb = (lane >> 4) << 4;
    const unsigned short* p = T + r * BT_RS + kb;
    FragU f;
    f.q[0] = *reinterpret_cast<const uint4*>(p);
    f.q[1] = *reinterpret_cast<const uint4*>(p + 8);
    return f.v;
}

// C/D 16x16 f32: VGPR j -> row m0 + j + 8*(lane>=16), col n0 + (lane&15)
__device__ __forceinline__ void store_c(float* C, int ld, int m0, int n0,
                                        int lane, const v8f& c) {
    int col   = n0 + (lane & 15);
    int rbase = m0 + ((lane >> 4) << 3);
#pragma unroll
    for (int j = 0; j < 8; ++j) C[(size_t)(rbase + j) * ld + col] = c[j];
}

// ---- TDM stage of a 128x32 bf16 weight tile into LDS (double-buffered) ----
// D# per CDNA5 ISA ch.8: group0 {count=1, lds_addr, global_addr, type=2};
// group1 {data_size=2B, pad 16B after each 64B row, tensor dims/stride, tile 32x128}.
__device__ __forceinline__ void stage_btile(unsigned short* dst,
                                            const unsigned short* W, int K, int N,
                                            int rowbase, int k0, int wave, int tid) {
#if HAVE_TDM
    if (wave == 0) {
        uint64_t ga = (uint64_t)(uintptr_t)(W + (size_t)rowbase * K + k0);
        v4u g0;
        g0[0] = 1u;                                           // count=1 (valid)
        g0[1] = (uint32_t)(uintptr_t)dst;                     // LDS byte address
        g0[2] = (uint32_t)ga;                                 // global_addr[31:0]
        g0[3] = (uint32_t)((ga >> 32) & 0x01FFFFFFu) | (2u << 30);  // addr[56:32] | type=2
        v8i_ g1;
        g1[0] = (int)((1u << 16) | (1u << 20) | (3u << 22) | (3u << 25));
                 // data_size=2B | pad_enable | pad_interval=16DW | pad_amount=4DW
        g1[1] = (int)(((unsigned)K & 0xFFFFu) << 16);         // tensor_dim0[15:0]
        g1[2] = (int)(((unsigned)K >> 16) & 0xFFFFu) |
                (int)(((unsigned)N & 0xFFFFu) << 16);         // td0[31:16] | td1[15:0]
        g1[3] = (int)(((unsigned)N >> 16) & 0xFFFFu) |
                (int)(32u << 16);                             // td1[31:16] | tile_dim0=32
        g1[4] = 128;                                          // tile_dim1=128, tile_dim2=0
        g1[5] = K;                                            // tensor_dim0_stride[31:0]
        g1[6] = 0;                                            // stride0[47:32] | stride1[15:0]
        g1[7] = 0;                                            // stride1 upper (unused, 2D)
        v4i_ z4 = {0, 0, 0, 0};
#if __clang_major__ >= 23
        v8i_ z8 = {0, 0, 0, 0, 0, 0, 0, 0};
        __builtin_amdgcn_tensor_load_to_lds(g0, g1, z4, z4, z8, 0);
#else
        __builtin_amdgcn_tensor_load_to_lds(g0, g1, z4, z4, 0);
#endif
    }
#else
    // cooperative fallback: 128 rows x 32 ushorts (one uint4 per row-quarter)
    for (int i = tid; i < 512; i += 256) {
        int r = i >> 2, c = i & 3;
        *reinterpret_cast<uint4*>(dst + r * BT_RS + c * 8) =
            *reinterpret_cast<const uint4*>(W + (size_t)(rowbase + r) * K + k0 + c * 8);
    }
#endif
    (void)tid; (void)N;
}

__device__ __forceinline__ void tile_wait(bool more, int wave) {
#if HAVE_TDM
    if (wave == 0) {
        if (more) __builtin_amdgcn_s_wait_tensorcnt((short)1);
        else      __builtin_amdgcn_s_wait_tensorcnt((short)0);
    }
#else
    (void)more; (void)wave;
#endif
}

// ---------------- small helper kernels ----------------
__global__ __launch_bounds__(256)
void zero_f32(float* p, int n) {
    int i = blockIdx.x * blockDim.x + threadIdx.x;
    if (i < n) p[i] = 0.f;
}

__global__ __launch_bounds__(256)
void pack_bf16(const float* __restrict__ src, unsigned short* __restrict__ dst,
               int rows, int scols, int dcols) {
    int idx = blockIdx.x * blockDim.x + threadIdx.x;
    if (idx >= rows * dcols) return;
    int r = idx / dcols, c = idx - r * dcols;
    float v = (c < scols) ? src[(size_t)r * scols + c] : 0.f;
    dst[idx] = f32_to_bf16(v);
}

__global__ __launch_bounds__(256)
void build_x(const float* __restrict__ sents, const float* __restrict__ mtab,
             const int* __restrict__ masks, unsigned short* __restrict__ xbf) {
    int idx = blockIdx.x * blockDim.x + threadIdx.x;
    if (idx >= BL * GINP) return;
    int bl = idx / GINP, c = idx - bl * GINP;
    float v;
    if (c < Dc)            v = sents[(size_t)bl * Dc + c];
    else if (c < Dc + MDc) v = mtab[(size_t)masks[bl] * MDc + (c - Dc)];
    else                   v = 0.f;
    xbf[idx] = f32_to_bf16(v);
}

// ---------------- big WMMA GEMM: C[M,N] = A[M,K] * W[N,K]^T (+bias[N]) ----
// block = 256 threads (8 waves, 4x2); block tile 64x128; wave tile 16x64.
// B tiles DMA'd to LDS by the Tensor Data Mover, double-buffered on TENSORcnt.
__global__ __launch_bounds__(256)
void wmma_gemm_nt(const unsigned short* __restrict__ A,
                  const unsigned short* __restrict__ W,
                  const float* __restrict__ bias,
                  float* __restrict__ C, int M, int N, int K) {
    __shared__ unsigned short btile[2][128 * BT_RS];

    int tid  = threadIdx.x;
    int lane = tid & 31;
    int wave = tid >> 5;
    int wr   = wave & 3;
    int wc   = wave >> 2;
    int m0      = blockIdx.y * 64 + wr * 16;
    int rowbase = blockIdx.x * 128;       // N offset of this block's B tile
    int nrel    = wc * 64;                // wave's N offset inside the tile

    v8f acc[4];
#pragma unroll
    for (int t = 0; t < 4; ++t) {
        float bv = bias ? bias[rowbase + nrel + 16 * t + (lane & 15)] : 0.f;
#pragma unroll
        for (int j = 0; j < 8; ++j) acc[t][j] = bv;
    }

    const int nk = K / 32;
    stage_btile(btile[0], W, K, N, rowbase, 0, wave, tid);
    for (int kt = 0; kt < nk; ++kt) {
        int buf = kt & 1;
        if (kt + 1 < nk)
            stage_btile(btile[buf ^ 1], W, K, N, rowbase, (kt + 1) * 32, wave, tid);
        tile_wait(kt + 1 < nk, wave);
        __syncthreads();

        v16bf a = load_frag_a(A, K, m0, kt * 32, lane);
#pragma unroll
        for (int t = 0; t < 4; ++t) {
            v16bf b = load_frag_b_lds(btile[buf], nrel + 16 * t, lane);
            acc[t] = __builtin_amdgcn_wmma_f32_16x16x32_bf16(
                false, a, false, b, (short)0, acc[t], false, false);
        }
        __syncthreads();
    }
#pragma unroll
    for (int t = 0; t < 4; ++t)
        store_c(C, N, m0, rowbase + nrel + 16 * t, lane, acc[t]);
    (void)M;
}

// ---------------- GRU recurrence ----------------
// grid = 16 blocks: blockIdx.x = dir*8 + slice ; each block owns 16 batch rows.
__global__ __launch_bounds__(256)
void gru_scan(const unsigned short* __restrict__ whhf_bf,
              const unsigned short* __restrict__ whhb_bf,
              const float* __restrict__ gi_f, const float* __restrict__ gi_b,
              const float* __restrict__ bihf, const float* __restrict__ bhhf,
              const float* __restrict__ bihb, const float* __restrict__ bhhb,
              const int* __restrict__ lens,
              float* __restrict__ hf_out, float* __restrict__ hbk_out) {
    __shared__ float          h32[16 * HDc];
    __shared__ unsigned short hbf[16 * HDc];
    __shared__ float          ghs[16 * G3];
    __shared__ float          s_bih[G3];
    __shared__ float          s_bhh[G3];
    __shared__ int            s_len[16];

    const int tid   = threadIdx.x;
    const int lane  = tid & 31;
    const int wave  = tid >> 5;
    const int dir   = blockIdx.x >> 3;
    const int slice = blockIdx.x & 7;
    const int bbase = slice * 16;

    const unsigned short* whh = dir ? whhb_bf : whhf_bf;
    const float* gi  = dir ? gi_b  : gi_f;
    const float* bih = dir ? bihb  : bihf;
    const float* bhh = dir ? bhhb  : bhhf;

    for (int i = tid; i < 16 * HDc; i += 256) { h32[i] = 0.f; hbf[i] = 0; }
    for (int i = tid; i < G3; i += 256) { s_bih[i] = bih[i]; s_bhh[i] = bhh[i]; }
    if (tid < 16) s_len[tid] = lens[bbase + tid];
    __syncthreads();

    const int nbase = wave * 96;       // 6 N-tiles of 16 per wave, 8 waves = 768
    const int me    = tid >> 4;        // elementwise row 0..15
    const int cb    = tid & 15;        // elementwise col base

    for (int t = 0; t < Lc; ++t) {
        // ---- gh = h @ Whh^T : M=16, N=768, K=256 ----
        v8f acc[6];
#pragma unroll
        for (int u = 0; u < 6; ++u)
#pragma unroll
            for (int j = 0; j < 8; ++j) acc[u][j] = 0.f;
        for (int k0 = 0; k0 < HDc; k0 += 32) {
            v16bf a = load_frag_a(hbf, HDc, 0, k0, lane);
#pragma unroll
            for (int u = 0; u < 6; ++u) {
                v16bf b = load_frag_b(whh, HDc, nbase + 16 * u, k0, lane);
                acc[u] = __builtin_amdgcn_wmma_f32_16x16x32_bf16(
                    false, a, false, b, (short)0, acc[u], false, false);
            }
        }
#pragma unroll
        for (int u = 0; u < 6; ++u) store_c(ghs, G3, 0, nbase + 16 * u, lane, acc[u]);
        __syncthreads();

        // ---- fused GRU cell ----
        const int  b      = bbase + me;
        const int  len    = s_len[me];
        const bool mvalid = (t < len);
        int pos = dir ? (len - 1 - t) : t;
        if (pos < 0) pos = 0;
        const float* gir = gi + ((size_t)b * Lc + pos) * G3;

        // prefetch next step's gi row (global_prefetch_b8)
        if (t + 1 < Lc) {
            int posn = dir ? (len - 2 - t) : (t + 1);
            if (posn < 0) posn = 0;
            __builtin_prefetch(gi + ((size_t)b * Lc + posn) * G3 + cb * 16, 0, 1);
        }

#pragma unroll
        for (int c = 0; c < 16; ++c) {
            int   i     = cb + (c << 4);       // 0..255
            float hprev = h32[me * HDc + i];
            float ir = gir[i]           + s_bih[i];
            float iz = gir[HDc + i]     + s_bih[HDc + i];
            float in_= gir[2 * HDc + i] + s_bih[2 * HDc + i];
            float hr = ghs[me * G3 + i]           + s_bhh[i];
            float hz = ghs[me * G3 + HDc + i]     + s_bhh[HDc + i];
            float hn = ghs[me * G3 + 2 * HDc + i] + s_bhh[2 * HDc + i];
            float r = 1.f / (1.f + expf(-(ir + hr)));
            float z = 1.f / (1.f + expf(-(iz + hz)));
            float n = tanhf(in_ + r * hn);
            float hnew = (1.f - z) * n + z * hprev;
            float hc   = mvalid ? hnew : hprev;
            h32[me * HDc + i] = hc;
            hbf[me * HDc + i] = f32_to_bf16(hc);
            if (dir == 0) {
                hf_out[((size_t)b * Lc + t) * HDc + i] = mvalid ? hnew : 0.f;
            } else if (mvalid) {
                hbk_out[((size_t)b * Lc + pos) * HDc + i] = hnew;   // hbk pre-zeroed
            }
        }
        __syncthreads();
    }
}

// ---------------- ctx = concat(hf,hbk) * pos_weight ; also bf16 copy -------
__global__ __launch_bounds__(256)
void ctx_kernel(const float* __restrict__ hf, const float* __restrict__ hbk,
                const int* __restrict__ masks, const int* __restrict__ lens,
                float* __restrict__ ctx, unsigned short* __restrict__ ctxbf) {
    int b = blockIdx.x;
    __shared__ int   s_begin, s_tnum;
    __shared__ float s_w[Lc];
    if (threadIdx.x == 0) { s_begin = Lc; s_tnum = 0; }
    __syncthreads();
    if (threadIdx.x < Lc && masks[b * Lc + threadIdx.x] == 1) {
        atomicMin(&s_begin, (int)threadIdx.x);
        atomicAdd(&s_tnum, 1);
    }
    __syncthreads();
    int   begin = (s_tnum == 0) ? 0 : s_begin;
    int   tnum  = s_tnum;
    int   len   = lens[b];
    float lf    = (float)len;
    if (threadIdx.x < Lc) {
        int j = threadIdx.x;
        float w = 0.f;
        if (j < begin)              w = 1.f - (float)(begin - j) / lf;
        if (masks[b * Lc + j] == 1) w = 1.f;
        if (j > begin + tnum)       w = 1.f - (float)(j - begin) / lf;
        if (j > len)                w = 0.f;
        s_w[j] = w;
    }
    __syncthreads();
    for (int idx = threadIdx.x; idx < Lc * Hc; idx += blockDim.x) {
        int l = idx >> 9, h = idx & 511;
        float v = (h < HDc) ? hf[((size_t)b * Lc + l) * HDc + h]
                            : hbk[((size_t)b * Lc + l) * HDc + (h - HDc)];
        v *= s_w[l];
        ctx[((size_t)b * Lc + l) * Hc + h]   = v;
        ctxbf[((size_t)b * Lc + l) * Hc + h] = f32_to_bf16(v);
    }
}

// ---------------- fk = ck @ tW^T + tb (T=4, tiny) --------------------------
__global__ __launch_bounds__(256)
void feat2tri(const float* __restrict__ ck, const float* __restrict__ tW,
              const float* __restrict__ tb, float* __restrict__ fk) {
    int idx = blockIdx.x * blockDim.x + threadIdx.x;   // BL*16
    if (idx >= BL * 16) return;
    int t  = idx & 3;
    int k  = (idx >> 2) & 3;
    int bl = idx >> 4;
    const float* c = ck + (size_t)bl * 512 + k * Rc;
    const float* w = tW + (size_t)(k * Tc + t) * Rc;
    float s = tb[k * Tc + t];
#pragma unroll 8
    for (int r = 0; r < Rc; ++r) s += c[r] * w[r];
    int b = bl / Lc, l = bl - b * Lc;
    fk[(((size_t)k * Bc + b) * Lc + l) * Tc + t] = s;
}

__device__ __forceinline__ float lse4(float a, float b, float c, float d) {
    float m = fmaxf(fmaxf(a, b), fmaxf(c, d));
    return m + logf(expf(a - m) + expf(b - m) + expf(c - m) + expf(d - m));
}

// ---------------- CRF forward/backward, marginals, normalized sp -----------
__global__ __launch_bounds__(256)
void crf_chain(const float* __restrict__ fk, const float* __restrict__ trans,
               const int* __restrict__ lens, float* __restrict__ alphas,
               float* __restrict__ sp) {
    int c = blockIdx.x * blockDim.x + threadIdx.x;     // 0..511 = k*B + b
    if (c >= 4 * Bc) return;
    int k = c / Bc, b = c - k * Bc;
    const float* f  = fk + (size_t)c * Lc * Tc;
    const float* tr = trans + (size_t)k * 16;          // tr[i*4+t]
    float* al = alphas + (size_t)c * Lc * Tc;
    int len = lens[b];

    float a0 = f[0], a1 = f[1], a2 = f[2], a3 = f[3];
    al[0] = a0; al[1] = a1; al[2] = a2; al[3] = a3;
    for (int l = 1; l < Lc; ++l) {
        if (l < len) {
            float n0 = lse4(a0 + tr[0], a1 + tr[4],  a2 + tr[8],  a3 + tr[12]) + f[l * 4 + 0];
            float n1 = lse4(a0 + tr[1], a1 + tr[5],  a2 + tr[9],  a3 + tr[13]) + f[l * 4 + 1];
            float n2 = lse4(a0 + tr[2], a1 + tr[6],  a2 + tr[10], a3 + tr[14]) + f[l * 4 + 2];
            float n3 = lse4(a0 + tr[3], a1 + tr[7],  a2 + tr[11], a3 + tr[15]) + f[l * 4 + 3];
            a0 = n0; a1 = n1; a2 = n2; a3 = n3;
        }
        al[l * 4 + 0] = a0; al[l * 4 + 1] = a1; al[l * 4 + 2] = a2; al[l * 4 + 3] = a3;
    }

    float b0 = 0.f, b1 = 0.f, b2 = 0.f, b3 = 0.f, gsum = 0.f;
    for (int l = Lc - 1; l >= 0; --l) {
        float s0 = al[l * 4 + 0] + b0, s1 = al[l * 4 + 1] + b1;
        float s2 = al[l * 4 + 2] + b2, s3 = al[l * 4 + 3] + b3;
        float mx = fmaxf(fmaxf(s0, s1), fmaxf(s2, s3));
        float e0 = expf(s0 - mx), e1 = expf(s1 - mx);
        float e2 = expf(s2 - mx), e3 = expf(s3 - mx);
        float spl = (l < len) ? (e1 / (e0 + e1 + e2 + e3)) : 0.f;
        sp[(size_t)c * Lc + l] = spl;
        gsum += spl;
        if (l > 0 && l < len) {
            float g0 = f[l * 4 + 0] + b0, g1 = f[l * 4 + 1] + b1;
            float g2 = f[l * 4 + 2] + b2, g3 = f[l * 4 + 3] + b3;
            float n0 = lse4(tr[0]  + g0, tr[1]  + g1, tr[2]  + g2, tr[3]  + g3);
            float n1 = lse4(tr[4]  + g0, tr[5]  + g1, tr[6]  + g2, tr[7]  + g3);
            float n2 = lse4(tr[8]  + g0, tr[9]  + g1, tr[10] + g2, tr[11] + g3);
            float n3 = lse4(tr[12] + g0, tr[13] + g1, tr[14] + g2, tr[15] + g3);
            b0 = n0; b1 = n1; b2 = n2; b3 = n3;
        }
    }
    float inv = 1.f / gsum;
    for (int l = 0; l < Lc; ++l) sp[(size_t)c * Lc + l] *= inv;
}

// ---------------- weighted pooling + ReLU -> sv[B, 4H] ---------------------
__global__ __launch_bounds__(256)
void pool_kernel(const float* __restrict__ sp, const float* __restrict__ ctx,
                 float* __restrict__ sv) {
    int c = blockIdx.x;                 // k*B + b
    int k = c / Bc, b = c - k * Bc;
    int h = threadIdx.x;                // 256 threads -> h and h+256
    float acc0 = 0.f, acc1 = 0.f;
    for (int l = 0; l < Lc; ++l) {
        float s = sp[(size_t)c * Lc + l];
        const float* cp = ctx + ((size_t)b * Lc + l) * Hc;
        acc0 += s * cp[h];
        acc1 += s * cp[h + 256];
    }
    sv[(size_t)b * (4 * Hc) + k * Hc + h]       = fmaxf(acc0, 0.f);
    sv[(size_t)b * (4 * Hc) + k * Hc + h + 256] = fmaxf(acc1, 0.f);
}

// ---------------- classifier + log_softmax + mean NLL ----------------------
__global__ __launch_bounds__(128)
void final_kernel(const float* __restrict__ sv, const float* __restrict__ lW,
                  const float* __restrict__ lb, const int* __restrict__ labels,
                  float* __restrict__ out) {
    __shared__ float red[128];
    int b = threadIdx.x;
    float loss = 0.f;
    {
        const float* v = sv + (size_t)b * (4 * Hc);
        float s[3];
#pragma unroll
        for (int c = 0; c < 3; ++c) {
            const float* w = lW + (size_t)c * (4 * Hc);
            float acc = lb[c];
            for (int j = 0; j < 4 * Hc; ++j) acc += v[j] * w[j];
            s[c] = acc;
        }
        float mx  = fmaxf(s[0], fmaxf(s[1], s[2]));
        float lse = mx + logf(expf(s[0] - mx) + expf(s[1] - mx) + expf(s[2] - mx));
        loss = -(s[labels[b]] - lse);
    }
    red[b] = loss;
    __syncthreads();
    for (int st = 64; st > 0; st >>= 1) {
        if (b < st) red[b] += red[b + st];
        __syncthreads();
    }
    if (b == 0) out[0] = red[0] / (float)Bc;
}

// ---------------- host: launch sequence ------------------------------------
extern "C" void kernel_launch(void* const* d_in, const int* in_sizes, int n_in,
                              void* d_out, int out_size, void* d_ws, size_t ws_size,
                              hipStream_t stream) {
    (void)in_sizes; (void)n_in; (void)out_size; (void)ws_size;
    const float* sents = (const float*)d_in[0];
    const float* mtab  = (const float*)d_in[1];
    const float* gWihf = (const float*)d_in[2];
    const float* gWhhf = (const float*)d_in[3];
    const float* gbihf = (const float*)d_in[4];
    const float* gbhhf = (const float*)d_in[5];
    const float* gWihb = (const float*)d_in[6];
    const float* gWhhb = (const float*)d_in[7];
    const float* gbihb = (const float*)d_in[8];
    const float* gbhhb = (const float*)d_in[9];
    const float* hW    = (const float*)d_in[10];
    const float* hb    = (const float*)d_in[11];
    const float* tW    = (const float*)d_in[12];
    const float* tb    = (const float*)d_in[13];
    const float* trans = (const float*)d_in[14];
    const float* lW    = (const float*)d_in[15];
    const float* lb    = (const float*)d_in[16];
    const int*   masks = (const int*)d_in[17];
    const int*   lens  = (const int*)d_in[18];
    const int*   labels= (const int*)d_in[19];
    float* out = (float*)d_out;

    char* ws = (char*)d_ws;
    size_t off = 0;
    auto give = [&](size_t bytes) -> char* {
        char* p = ws + off;
        off = (off + bytes + 255) & ~(size_t)255;
        return p;
    };
    unsigned short* xbf     = (unsigned short*)give((size_t)BL * GINP * 2);
    unsigned short* wihf_bf = (unsigned short*)give((size_t)G3 * GINP * 2);
    unsigned short* wihb_bf = (unsigned short*)give((size_t)G3 * GINP * 2);
    unsigned short* whhf_bf = (unsigned short*)give((size_t)G3 * HDc * 2);
    unsigned short* whhb_bf = (unsigned short*)give((size_t)G3 * HDc * 2);
    unsigned short* hw_bf   = (unsigned short*)give((size_t)512 * 512 * 2);
    float* gi_f   = (float*)give((size_t)BL * G3 * 4);
    float* gi_b   = (float*)give((size_t)BL * G3 * 4);
    float* hf     = (float*)give((size_t)BL * HDc * 4);
    float* hbk    = (float*)give((size_t)BL * HDc * 4);
    float* ctx    = (float*)give((size_t)BL * Hc * 4);
    unsigned short* ctxbf = (unsigned short*)give((size_t)BL * Hc * 2);
    float* ck     = (float*)give((size_t)BL * 512 * 4);
    float* fk     = (float*)give((size_t)4 * BL * Tc * 4);
    float* alphas = (float*)give((size_t)4 * Bc * Lc * Tc * 4);
    float* sp     = (float*)give((size_t)4 * Bc * Lc * 4);
    float* sv     = (float*)give((size_t)Bc * 4 * Hc * 4);

    // 0) zero hbk (positions >= len are never written by the scan)
    zero_f32<<<(BL * HDc + 255) / 256, 256, 0, stream>>>(hbk, BL * HDc);

    // 1) inputs -> bf16
    build_x<<<(BL * GINP + 255) / 256, 256, 0, stream>>>(sents, mtab, masks, xbf);
    pack_bf16<<<(G3 * GINP + 255) / 256, 256, 0, stream>>>(gWihf, wihf_bf, G3, GINc, GINP);
    pack_bf16<<<(G3 * GINP + 255) / 256, 256, 0, stream>>>(gWihb, wihb_bf, G3, GINc, GINP);
    pack_bf16<<<(G3 * HDc + 255) / 256, 256, 0, stream>>>(gWhhf, whhf_bf, G3, HDc, HDc);
    pack_bf16<<<(G3 * HDc + 255) / 256, 256, 0, stream>>>(gWhhb, whhb_bf, G3, HDc, HDc);
    pack_bf16<<<(512 * 512 + 255) / 256, 256, 0, stream>>>(hW, hw_bf, 512, 512, 512);

    // 2) gi = x @ Wih^T (both directions), WMMA bf16 + TDM-staged B tiles
    {
        dim3 grid(G3 / 128, BL / 64);
        wmma_gemm_nt<<<grid, 256, 0, stream>>>(xbf, wihf_bf, nullptr, gi_f, BL, G3, GINP);
        wmma_gemm_nt<<<grid, 256, 0, stream>>>(xbf, wihb_bf, nullptr, gi_b, BL, G3, GINP);
    }

    // 3) bidirectional GRU scan (16 persistent workgroups)
    gru_scan<<<16, 256, 0, stream>>>(whhf_bf, whhb_bf, gi_f, gi_b,
                                     gbihf, gbhhf, gbihb, gbhhb, lens, hf, hbk);

    // 4) ctx = concat(hf, hbk) * pos_weights
    ctx_kernel<<<Bc, 256, 0, stream>>>(hf, hbk, masks, lens, ctx, ctxbf);

    // 5) ck = ctx @ hW^T + hb  (all 4 branches fused: N = 4*R = 512), WMMA
    {
        dim3 grid(512 / 128, BL / 64);
        wmma_gemm_nt<<<grid, 256, 0, stream>>>(ctxbf, hw_bf, hb, ck, BL, 512, 512);
    }

    // 6) fk = ck @ tW^T + tb
    feat2tri<<<(BL * 16 + 255) / 256, 256, 0, stream>>>(ck, tW, tb, fk);

    // 7) CRF marginals -> normalized sp
    crf_chain<<<2, 256, 0, stream>>>(fk, trans, lens, alphas, sp);

    // 8) pooling + ReLU
    pool_kernel<<<4 * Bc, 256, 0, stream>>>(sp, ctx, sv);

    // 9) classifier + mean NLL
    final_kernel<<<1, 128, 0, stream>>>(sv, lW, lb, labels, out);
}
